// DepthAwareGAT_15522011808331
// MI455X (gfx1250) — matrix operand order repaired and checked by reading, not compiled, not run on hardware
//
#include <hip/hip_runtime.h>
#include <math.h>

#define HID 32
#define HEADS 4
#define NCLS 5
#define EFEAT 18
#define HC 128           // HEADS*HID
#define NEG_SLOPE 0.2f

typedef __attribute__((ext_vector_type(16))) _Float16 v16h;
typedef __attribute__((ext_vector_type(8)))  _Float16 v8h;
typedef __attribute__((ext_vector_type(8)))  float    v8f;

// ---- monotone float<->uint encoding for atomicMax on floats ----
__device__ __forceinline__ unsigned fenc(float f) {
  unsigned u = __float_as_uint(f);
  return u ^ (((int)u >> 31) | 0x80000000u);
}
__device__ __forceinline__ float fdec(unsigned e) {
  unsigned mask = (e >> 31) ? 0x80000000u : 0xFFFFFFFFu;
  return __uint_as_float(e ^ mask);
}

// ============================================================
// Weight pre-pass: Wt[n][k] = (f16) W[k][n]  (tiny: <=64KB)
// ============================================================
__global__ void conv_w_f16t(const float* __restrict__ W, _Float16* __restrict__ Wt,
                            int K, int Ncol) {
  int idx = blockIdx.x * blockDim.x + threadIdx.x;
  if (idx >= K * Ncol) return;
  int k = idx / Ncol, n = idx % Ncol;
  Wt[(size_t)n * K + k] = (_Float16)W[idx];
}

// ============================================================
// WMMA GEMM: C[M,Ncol] = A[M,K] * B[K,Ncol].
// A f32 row-major (float4 fragment loads), B pre-transposed f16 [Ncol][K]
// (v8h fragment loads). One wave per 16x16 tile, K unrolled at compile time.
// ============================================================
template <int K>
__global__ void gemm_wmma_f16(const float* __restrict__ A, const _Float16* __restrict__ Bt,
                              float* __restrict__ C, int M, int Ncol) {
  const int lane   = threadIdx.x & 31;
  const int wave   = threadIdx.x >> 5;
  const int tilesN = Ncol >> 4;
  const int tile   = blockIdx.x * (blockDim.x >> 5) + wave;
  const int totTiles = (M >> 4) * tilesN;
  if (tile >= totTiles) return;             // wave-uniform: EXEC stays full
  const int m0 = (tile / tilesN) << 4;
  const int n0 = (tile % tilesN) << 4;
  const int half = lane >> 4;
  const int mr = lane & 15;                 // A row / B,C column within tile

  // A lane layout (16-bit A 16x32): lanes 0-15 K {0..7,16..23}, lanes 16-31 K {8..15,24..31}
  const float*    Arow = A  + (size_t)(m0 + mr) * K + half * 8;
  // B lane layout (K x 16): lane holds column n0+mr, K {0..7,16..23} (half 0) / {8..15,24..31} (half 1)
  const _Float16* Brow = Bt + (size_t)(n0 + mr) * K + half * 8;

  v8f acc = {};
#pragma unroll
  for (int k0 = 0; k0 < K; k0 += 32) {
    const float4 a0 = *(const float4*)(Arow + k0);
    const float4 a1 = *(const float4*)(Arow + k0 + 4);
    const float4 a2 = *(const float4*)(Arow + k0 + 16);
    const float4 a3 = *(const float4*)(Arow + k0 + 20);
    const v8h    b0 = *(const v8h*)(Brow + k0);
    const v8h    b1 = *(const v8h*)(Brow + k0 + 16);
    v16h a, b;
    a[0]  = (_Float16)a0.x; a[1]  = (_Float16)a0.y; a[2]  = (_Float16)a0.z; a[3]  = (_Float16)a0.w;
    a[4]  = (_Float16)a1.x; a[5]  = (_Float16)a1.y; a[6]  = (_Float16)a1.z; a[7]  = (_Float16)a1.w;
    a[8]  = (_Float16)a2.x; a[9]  = (_Float16)a2.y; a[10] = (_Float16)a2.z; a[11] = (_Float16)a2.w;
    a[12] = (_Float16)a3.x; a[13] = (_Float16)a3.y; a[14] = (_Float16)a3.z; a[15] = (_Float16)a3.w;
#pragma unroll
    for (int j = 0; j < 8; ++j) { b[j] = b0[j]; b[j + 8] = b1[j]; }
    acc = __builtin_amdgcn_wmma_f32_16x16x32_f16(false, a, false, b, (short)0, acc, false, false);
  }
  // C/D layout: VGPR r -> row m0 + r + 8*half, col n0 + (lane&15)
#pragma unroll
  for (int r = 0; r < 8; ++r)
    C[(size_t)(m0 + r + 8 * half) * Ncol + n0 + mr] = acc[r];
}

// Scalar tail for M not divisible by 16 (not hit for N=100000, kept for safety)
__global__ void gemm_tail(const float* __restrict__ A, const float* __restrict__ B,
                          float* __restrict__ C, int mStart, int M, int K, int Ncol) {
  int idx = blockIdx.x * blockDim.x + threadIdx.x;
  int rows = M - mStart;
  if (idx >= rows * Ncol) return;
  int r = mStart + idx / Ncol, c = idx % Ncol;
  float s = 0.f;
  for (int k = 0; k < K; ++k) s += A[(size_t)r * K + k] * B[(size_t)k * Ncol + c];
  C[(size_t)r * Ncol + c] = s;
}

// ============================================================
// Edge encoder MLP: relu(ea@ew1+eb1)@ew2+eb2 -> ee[E,4]
// ============================================================
__global__ void edge_encoder_kernel(const float* __restrict__ ea,
                                    const float* __restrict__ ew1, const float* __restrict__ eb1,
                                    const float* __restrict__ ew2, const float* __restrict__ eb2,
                                    float* __restrict__ ee, int E) {
  __shared__ float sW1[EFEAT * HID];
  __shared__ float sB1[HID];
  __shared__ float sW2[HID * HEADS];
  __shared__ float sB2[HEADS];
  for (int i = threadIdx.x; i < EFEAT * HID; i += blockDim.x) sW1[i] = ew1[i];
  for (int i = threadIdx.x; i < HID; i += blockDim.x) sB1[i] = eb1[i];
  for (int i = threadIdx.x; i < HID * HEADS; i += blockDim.x) sW2[i] = ew2[i];
  for (int i = threadIdx.x; i < HEADS; i += blockDim.x) sB2[i] = eb2[i];
  __syncthreads();
  int e = blockIdx.x * blockDim.x + threadIdx.x;
  if (e >= E) return;
  float av[EFEAT];
#pragma unroll
  for (int i = 0; i < EFEAT; ++i) av[i] = ea[(size_t)e * EFEAT + i];
  float t[HID];
#pragma unroll
  for (int j = 0; j < HID; ++j) {
    float s = sB1[j];
#pragma unroll
    for (int i = 0; i < EFEAT; ++i) s += av[i] * sW1[i * HID + j];
    t[j] = s > 0.f ? s : 0.f;
  }
#pragma unroll
  for (int h = 0; h < HEADS; ++h) {
    float s = sB2[h];
#pragma unroll
    for (int j = 0; j < HID; ++j) s += t[j] * sW2[j * HEADS + h];
    ee[(size_t)e * HEADS + h] = s;
  }
}

// ============================================================
// Precompute collapsed edge-attention matrices:
// wae[k][h] = sum_c We[k, h*HID+c] * ae[h, c]   (layers 1,2)
// wae3[k]   = sum_c we3[k, c] * ae3[c]          (layer 3)
// ============================================================
__global__ void precompute_wae(const float* __restrict__ we1, const float* __restrict__ ae1,
                               const float* __restrict__ we2, const float* __restrict__ ae2,
                               const float* __restrict__ we3, const float* __restrict__ ae3,
                               float* __restrict__ wae1, float* __restrict__ wae2,
                               float* __restrict__ wae3) {
  int t = threadIdx.x;
  if (t < HEADS * HEADS) {
    int k = t >> 2, h = t & 3;
    float s1 = 0.f, s2 = 0.f;
    for (int c = 0; c < HID; ++c) {
      s1 += we1[k * HC + h * HID + c] * ae1[h * HID + c];
      s2 += we2[k * HC + h * HID + c] * ae2[h * HID + c];
    }
    wae1[t] = s1; wae2[t] = s2;
  }
  if (t < HEADS) {
    float s = 0.f;
    for (int c = 0; c < NCLS; ++c) s += we3[t * NCLS + c] * ae3[c];
    wae3[t] = s;
  }
}

// ============================================================
// Per-(node,head): attention dots + init softmax state + zero output
// ============================================================
__global__ void node_prep(const float* __restrict__ h, const float* __restrict__ att_s,
                          const float* __restrict__ att_d,
                          float* __restrict__ asrc, float* __restrict__ adst,
                          unsigned* __restrict__ amaxu, float* __restrict__ denom,
                          float* __restrict__ outbuf, int n) {
  int idx = blockIdx.x * blockDim.x + threadIdx.x;
  if (idx >= n * HEADS) return;
  int node = idx >> 2, hh = idx & 3;
  const float* row = h + (size_t)node * HC + hh * HID;
  float s = 0.f, d = 0.f;
#pragma unroll
  for (int c = 0; c < HID; ++c) { s += row[c] * att_s[hh * HID + c]; d += row[c] * att_d[hh * HID + c]; }
  asrc[idx] = s; adst[idx] = d;
  amaxu[idx] = 0u; denom[idx] = 0.f;
  float* o = outbuf + (size_t)node * HC + hh * HID;
#pragma unroll
  for (int c = 0; c < HID; ++c) o[c] = 0.f;
}

// ============================================================
// Edge pass 1: alpha = leaky_relu(as[src]+ad[dst]+a_e), segment max
// ============================================================
__global__ void edge_alpha_max(const int* __restrict__ src, const int* __restrict__ dst,
                               const float* __restrict__ ee, const float* __restrict__ wae,
                               const float* __restrict__ asrc, const float* __restrict__ adst,
                               float* __restrict__ alphabuf, unsigned* __restrict__ amaxu, int E) {
  int idx = blockIdx.x * blockDim.x + threadIdx.x;
  if (idx >= E * HEADS) return;
  int e = idx >> 2, hh = idx & 3;
  int s = src[e], d = dst[e];
  const float4 ev = *(const float4*)(ee + (size_t)e * HEADS);
  float ae = ev.x * wae[0 * HEADS + hh] + ev.y * wae[1 * HEADS + hh] +
             ev.z * wae[2 * HEADS + hh] + ev.w * wae[3 * HEADS + hh];
  float a = asrc[s * HEADS + hh] + adst[d * HEADS + hh] + ae;
  a = a > 0.f ? a : NEG_SLOPE * a;
  alphabuf[idx] = a;
  atomicMax(&amaxu[d * HEADS + hh], fenc(a));
}

// Edge pass 2: exp(alpha - max), segment sum
__global__ void edge_exp_sum(const int* __restrict__ dst, float* __restrict__ alphabuf,
                             const unsigned* __restrict__ amaxu, float* __restrict__ denom, int E) {
  int idx = blockIdx.x * blockDim.x + threadIdx.x;
  if (idx >= E * HEADS) return;
  int e = idx >> 2, hh = idx & 3;
  int d = dst[e];
  float a = expf(alphabuf[idx] - fdec(amaxu[d * HEADS + hh]));
  alphabuf[idx] = a;
  atomicAdd(&denom[d * HEADS + hh], a);
}

// Edge pass 3: out[dst] += h[src] * alpha/denom. One wave per edge, 4 channels/lane.
__global__ void edge_aggregate(const int* __restrict__ src, const int* __restrict__ dst,
                               const float* __restrict__ alphabuf, const float* __restrict__ denom,
                               const float* __restrict__ h, float* __restrict__ outbuf, int E) {
  int gid = blockIdx.x * blockDim.x + threadIdx.x;
  int e = gid >> 5;
  if (e >= E) return;
  int lane = gid & 31;
  int s = src[e], d = dst[e];
  const float4 al = *(const float4*)(alphabuf + (size_t)e * HEADS);
  const float4 dn = *(const float4*)(denom + (size_t)d * HEADS);
  float w[HEADS] = { al.x / (dn.x + 1e-16f), al.y / (dn.y + 1e-16f),
                     al.z / (dn.z + 1e-16f), al.w / (dn.w + 1e-16f) };
  const float* hs = h + (size_t)s * HC;
  float* od = outbuf + (size_t)d * HC;
#pragma unroll
  for (int i = 0; i < HEADS; ++i) {
    int c = lane + 32 * i;   // head = i
    atomicAdd(&od[c], hs[c] * w[i]);
  }
}

// out = elu(out + bias), in place
__global__ void bias_elu(float* __restrict__ buf, const float* __restrict__ bias, int n) {
  int idx = blockIdx.x * blockDim.x + threadIdx.x;
  if (idx >= n * HC) return;
  float v = buf[idx] + bias[idx & (HC - 1)];
  buf[idx] = v > 0.f ? v : (expf(v) - 1.f);
}

// ============================================================
// Layer 3 (heads=1, out=NCLS): per-node projection + dots + init
// ============================================================
__global__ void node3_prep(const float* __restrict__ hin, const float* __restrict__ w3,
                           const float* __restrict__ as3, const float* __restrict__ ad3,
                           float* __restrict__ h3, float* __restrict__ asrc3,
                           float* __restrict__ adst3, unsigned* __restrict__ amaxu3,
                           float* __restrict__ denom3, float* __restrict__ out3, int n) {
  int node = blockIdx.x * blockDim.x + threadIdx.x;
  if (node >= n) return;
  const float* row = hin + (size_t)node * HC;
  float acc[NCLS] = {0.f, 0.f, 0.f, 0.f, 0.f};
  for (int k = 0; k < HC; ++k) {
    float x = row[k];
#pragma unroll
    for (int c = 0; c < NCLS; ++c) acc[c] += x * w3[k * NCLS + c];
  }
  float s = 0.f, d = 0.f;
#pragma unroll
  for (int c = 0; c < NCLS; ++c) {
    h3[(size_t)node * NCLS + c] = acc[c];
    out3[(size_t)node * NCLS + c] = 0.f;
    s += acc[c] * as3[c];
    d += acc[c] * ad3[c];
  }
  asrc3[node] = s; adst3[node] = d;
  amaxu3[node] = 0u; denom3[node] = 0.f;
}

__global__ void edge3_alpha_max(const int* __restrict__ src, const int* __restrict__ dst,
                                const float* __restrict__ ee, const float* __restrict__ wae3,
                                const float* __restrict__ asrc3, const float* __restrict__ adst3,
                                float* __restrict__ alphabuf, unsigned* __restrict__ amaxu3, int E) {
  int e = blockIdx.x * blockDim.x + threadIdx.x;
  if (e >= E) return;
  const float4 ev = *(const float4*)(ee + (size_t)e * HEADS);
  float ae = ev.x * wae3[0] + ev.y * wae3[1] + ev.z * wae3[2] + ev.w * wae3[3];
  int d = dst[e];
  float a = asrc3[src[e]] + adst3[d] + ae;
  a = a > 0.f ? a : NEG_SLOPE * a;
  alphabuf[e] = a;
  atomicMax(&amaxu3[d], fenc(a));
}

__global__ void edge3_exp_sum(const int* __restrict__ dst, float* __restrict__ alphabuf,
                              const unsigned* __restrict__ amaxu3, float* __restrict__ denom3, int E) {
  int e = blockIdx.x * blockDim.x + threadIdx.x;
  if (e >= E) return;
  int d = dst[e];
  float a = expf(alphabuf[e] - fdec(amaxu3[d]));
  alphabuf[e] = a;
  atomicAdd(&denom3[d], a);
}

__global__ void edge3_aggregate(const int* __restrict__ src, const int* __restrict__ dst,
                                const float* __restrict__ alphabuf, const float* __restrict__ denom3,
                                const float* __restrict__ h3, float* __restrict__ out3, int E) {
  int e = blockIdx.x * blockDim.x + threadIdx.x;
  if (e >= E) return;
  int s = src[e], d = dst[e];
  float w = alphabuf[e] / (denom3[d] + 1e-16f);
#pragma unroll
  for (int c = 0; c < NCLS; ++c)
    atomicAdd(&out3[(size_t)d * NCLS + c], h3[(size_t)s * NCLS + c] * w);
}

__global__ void log_softmax_out(const float* __restrict__ out3, const float* __restrict__ b3,
                                float* __restrict__ dout, int n) {
  int node = blockIdx.x * blockDim.x + threadIdx.x;
  if (node >= n) return;
  float x[NCLS];
  float m = -3.0e38f;
#pragma unroll
  for (int c = 0; c < NCLS; ++c) { x[c] = out3[(size_t)node * NCLS + c] + b3[c]; m = fmaxf(m, x[c]); }
  float sum = 0.f;
#pragma unroll
  for (int c = 0; c < NCLS; ++c) sum += expf(x[c] - m);
  float l = m + logf(sum);
#pragma unroll
  for (int c = 0; c < NCLS; ++c) dout[(size_t)node * NCLS + c] = x[c] - l;
}

// ============================================================
extern "C" void kernel_launch(void* const* d_in, const int* in_sizes, int n_in,
                              void* d_out, int out_size, void* d_ws, size_t ws_size,
                              hipStream_t stream) {
  const float* x   = (const float*)d_in[0];
  const int*   ei  = (const int*)d_in[1];
  const float* ea  = (const float*)d_in[2];
  const float* ew1 = (const float*)d_in[3];
  const float* eb1 = (const float*)d_in[4];
  const float* ew2 = (const float*)d_in[5];
  const float* eb2 = (const float*)d_in[6];
  const float* w1  = (const float*)d_in[7];
  const float* as1 = (const float*)d_in[8];
  const float* ad1 = (const float*)d_in[9];
  const float* we1 = (const float*)d_in[10];
  const float* ae1 = (const float*)d_in[11];
  const float* b1  = (const float*)d_in[12];
  const float* w2  = (const float*)d_in[13];
  const float* as2 = (const float*)d_in[14];
  const float* ad2 = (const float*)d_in[15];
  const float* we2 = (const float*)d_in[16];
  const float* ae2 = (const float*)d_in[17];
  const float* b2  = (const float*)d_in[18];
  const float* w3  = (const float*)d_in[19];
  const float* as3 = (const float*)d_in[20];
  const float* ad3 = (const float*)d_in[21];
  const float* we3 = (const float*)d_in[22];
  const float* ae3 = (const float*)d_in[23];
  const float* b3  = (const float*)d_in[24];

  const int F_IN = 64;
  const int N = in_sizes[0] / F_IN;
  const int E = in_sizes[2] / EFEAT;
  const int* src = ei;
  const int* dst = ei + E;

  // ---- workspace layout (float slots; all section sizes are multiples of 4 -> 16B aligned) ----
  float* wsf = (float*)d_ws;
  size_t o = 0;
  float*    ee_buf = wsf + o; o += (size_t)E * HEADS;
  float*    alphab = wsf + o; o += (size_t)E * HEADS;
  float*    hA     = wsf + o; o += (size_t)N * HC;       // projected features
  float*    hB     = wsf + o; o += (size_t)N * HC;       // aggregate / layer input
  float*    asrcb  = wsf + o; o += (size_t)N * HEADS;
  float*    adstb  = wsf + o; o += (size_t)N * HEADS;
  unsigned* amaxu  = (unsigned*)(wsf + o); o += (size_t)N * HEADS;
  float*    denomb = wsf + o; o += (size_t)N * HEADS;
  float*    h3b    = wsf + o; o += ((size_t)N * NCLS + 3) & ~(size_t)3;
  float*    out3b  = wsf + o; o += ((size_t)N * NCLS + 3) & ~(size_t)3;
  float*    asrc3  = wsf + o; o += ((size_t)N + 3) & ~(size_t)3;
  float*    adst3  = wsf + o; o += ((size_t)N + 3) & ~(size_t)3;
  unsigned* amax3  = (unsigned*)(wsf + o); o += ((size_t)N + 3) & ~(size_t)3;
  float*    denom3 = wsf + o; o += ((size_t)N + 3) & ~(size_t)3;
  float*    wae1b  = wsf + o; o += 16;
  float*    wae2b  = wsf + o; o += 16;
  float*    wae3b  = wsf + o; o += 16;
  _Float16* w1t    = (_Float16*)(wsf + o); o += (size_t)(F_IN * HC) / 2;  // f16 halves in float slots
  _Float16* w2t    = (_Float16*)(wsf + o); o += (size_t)(HC * HC) / 2;

  const int TB = 256;
  auto blks = [](long long n, int tb) { return (int)((n + tb - 1) / tb); };

  // 0) tiny precomputes + edge encoder (shared across layers)
  precompute_wae<<<1, 32, 0, stream>>>(we1, ae1, we2, ae2, we3, ae3, wae1b, wae2b, wae3b);
  conv_w_f16t<<<blks(F_IN * HC, TB), TB, 0, stream>>>(w1, w1t, F_IN, HC);
  conv_w_f16t<<<blks(HC * HC, TB), TB, 0, stream>>>(w2, w2t, HC, HC);
  edge_encoder_kernel<<<blks(E, TB), TB, 0, stream>>>(ea, ew1, eb1, ew2, eb2, ee_buf, E);

  // ---- layer 1: x[N,64] -> hB[N,128] ----
  {
    const int M16 = N & ~15;
    const int tiles = (M16 >> 4) * (HC >> 4);
    gemm_wmma_f16<64><<<blks(tiles, 8), TB, 0, stream>>>(x, w1t, hA, M16, HC);
    if (N != M16)
      gemm_tail<<<blks((long long)(N - M16) * HC, TB), TB, 0, stream>>>(x, w1, hA, M16, N, F_IN, HC);
    node_prep<<<blks((long long)N * HEADS, TB), TB, 0, stream>>>(hA, as1, ad1, asrcb, adstb, amaxu, denomb, hB, N);
    edge_alpha_max<<<blks((long long)E * HEADS, TB), TB, 0, stream>>>(src, dst, ee_buf, wae1b, asrcb, adstb, alphab, amaxu, E);
    edge_exp_sum<<<blks((long long)E * HEADS, TB), TB, 0, stream>>>(dst, alphab, amaxu, denomb, E);
    edge_aggregate<<<blks((long long)E * 32, TB), TB, 0, stream>>>(src, dst, alphab, denomb, hA, hB, E);
    bias_elu<<<blks((long long)N * HC, TB), TB, 0, stream>>>(hB, b1, N);
  }

  // ---- layer 2: hB[N,128] -> hB[N,128] ----
  {
    const int M16 = N & ~15;
    const int tiles = (M16 >> 4) * (HC >> 4);
    gemm_wmma_f16<128><<<blks(tiles, 8), TB, 0, stream>>>(hB, w2t, hA, M16, HC);
    if (N != M16)
      gemm_tail<<<blks((long long)(N - M16) * HC, TB), TB, 0, stream>>>(hB, w2, hA, M16, N, HC, HC);
    node_prep<<<blks((long long)N * HEADS, TB), TB, 0, stream>>>(hA, as2, ad2, asrcb, adstb, amaxu, denomb, hB, N);
    edge_alpha_max<<<blks((long long)E * HEADS, TB), TB, 0, stream>>>(src, dst, ee_buf, wae2b, asrcb, adstb, alphab, amaxu, E);
    edge_exp_sum<<<blks((long long)E * HEADS, TB), TB, 0, stream>>>(dst, alphab, amaxu, denomb, E);
    edge_aggregate<<<blks((long long)E * 32, TB), TB, 0, stream>>>(src, dst, alphab, denomb, hA, hB, E);
    bias_elu<<<blks((long long)N * HC, TB), TB, 0, stream>>>(hB, b2, N);
  }

  // ---- layer 3: hB[N,128] -> d_out[N,5] log_softmax ----
  node3_prep<<<blks(N, TB), TB, 0, stream>>>(hB, w3, as3, ad3, h3b, asrc3, adst3, amax3, denom3, out3b, N);
  edge3_alpha_max<<<blks(E, TB), TB, 0, stream>>>(src, dst, ee_buf, wae3b, asrc3, adst3, alphab, amax3, E);
  edge3_exp_sum<<<blks(E, TB), TB, 0, stream>>>(dst, alphab, amax3, denom3, E);
  edge3_aggregate<<<blks(E, TB), TB, 0, stream>>>(src, dst, alphab, denom3, h3b, out3b, E);
  log_softmax_out<<<blks(N, TB), TB, 0, stream>>>(out3b, b3, (float*)d_out, N);
}
